// MolSets_86672440033884
// MI455X (gfx1250) — matrix operand order repaired and checked by reading, not compiled, not run on hardware
//
#include <hip/hip_runtime.h>
#include <hip/hip_bf16.h>

typedef __attribute__((ext_vector_type(16))) _Float16 v16h;
typedef __attribute__((ext_vector_type(8)))  float    v8f;

#define TPB 256

// ---------------------------------------------------------------- utilities
__global__ __launch_bounds__(TPB) void zero_f32_k(float* p, long long n) {
  long long stride = (long long)gridDim.x * blockDim.x;
  for (long long i = (long long)blockIdx.x * blockDim.x + threadIdx.x; i < n; i += stride)
    p[i] = 0.f;
}

__global__ __launch_bounds__(TPB) void cvt_f32_f16_k(const float* __restrict__ in,
                                                     _Float16* __restrict__ out, long long n) {
  long long stride = (long long)gridDim.x * blockDim.x;
  for (long long i = (long long)blockIdx.x * blockDim.x + threadIdx.x; i < n; i += stride)
    out[i] = (_Float16)in[i];
}

__global__ __launch_bounds__(TPB) void degree_k(const int* __restrict__ dst, float* cnt, long long E) {
  long long stride = (long long)gridDim.x * blockDim.x;
  for (long long e = (long long)blockIdx.x * blockDim.x + threadIdx.x; e < E; e += stride)
    atomicAdd(&cnt[dst[e]], 1.f);
}

// scatter-add neighbor features (f32 source)
__global__ __launch_bounds__(TPB) void scatter_f32_k(const float* __restrict__ X,
                                                     const int* __restrict__ src,
                                                     const int* __restrict__ dst,
                                                     float* __restrict__ Agg, long long E, int C) {
  long long n = E * (long long)C;
  long long stride = (long long)gridDim.x * blockDim.x;
  for (long long i = (long long)blockIdx.x * blockDim.x + threadIdx.x; i < n; i += stride) {
    long long e = i / C; int c = (int)(i - e * C);
    atomicAdd(&Agg[(long long)dst[e] * C + c], X[(long long)src[e] * C + c]);
  }
}

// scatter-add neighbor features (f16 source)
__global__ __launch_bounds__(TPB) void scatter_f16_k(const _Float16* __restrict__ X,
                                                     const int* __restrict__ src,
                                                     const int* __restrict__ dst,
                                                     float* __restrict__ Agg, long long E, int C) {
  long long n = E * (long long)C;
  long long stride = (long long)gridDim.x * blockDim.x;
  for (long long i = (long long)blockIdx.x * blockDim.x + threadIdx.x; i < n; i += stride) {
    long long e = i / C; int c = (int)(i - e * C);
    atomicAdd(&Agg[(long long)dst[e] * C + c], (float)X[(long long)src[e] * C + c]);
  }
}

__global__ __launch_bounds__(TPB) void agg_div_k(float* __restrict__ Agg,
                                                 const float* __restrict__ cnt,
                                                 long long M, int C) {
  long long n = M * (long long)C;
  long long stride = (long long)gridDim.x * blockDim.x;
  for (long long i = (long long)blockIdx.x * blockDim.x + threadIdx.x; i < n; i += stride)
    Agg[i] /= fmaxf(cnt[i / C], 1.f);
}

// ---------------------------------------------------------------- WMMA SAGE GEMM
// Out[m,n] = sum_k Xf16[m,k]*Wr[n,k] + sum_k Agg[m,k]*Wl[n,k] + bias[n]
// A fragment (16x32 f16, ISA 7.12.2): lane l: row = tileM + (l&15); half=l>>4
//   elems 0..7  -> K = k0 + half*8 + e
//   elems 8..15 -> K = k0 + 16 + half*8 + (e-8)
__device__ __forceinline__ v16h load_a_f16(const _Float16* A, int lda, int m, int k0, int half) {
  const _Float16* p0 = A + (long long)m * lda + k0 + half * 8;
  const _Float16* p1 = p0 + 16;
  v16h a;
#pragma unroll
  for (int e = 0; e < 8; ++e) a[e] = p0[e];
#pragma unroll
  for (int e = 0; e < 8; ++e) a[8 + e] = p1[e];
  return a;
}
__device__ __forceinline__ v16h load_a_f32(const float* A, int lda, int m, int k0, int half) {
  const float* p0 = A + (long long)m * lda + k0 + half * 8;
  const float* p1 = p0 + 16;
  v16h a;
#pragma unroll
  for (int e = 0; e < 8; ++e) a[e] = (_Float16)p0[e];
#pragma unroll
  for (int e = 0; e < 8; ++e) a[8 + e] = (_Float16)p1[e];
  return a;
}
// B fragment (32x16 f16): lane l: col n = tileN + (l&15); half=l>>4
//   elems 0..7 -> K = k0 + half*8 + e; elems 8..15 -> K = k0 + 16 + half*8 + (e-8)  (W stored [N,K] row-major == B col-major)
__device__ __forceinline__ v16h load_b_f16(const _Float16* W, int K, int n, int k0, int half) {
  const _Float16* p = W + (long long)n * K + k0 + half * 8;
  v16h b;
#pragma unroll
  for (int e = 0; e < 8; ++e) b[e] = p[e];
#pragma unroll
  for (int e = 0; e < 8; ++e) b[8 + e] = p[16 + e];
  return b;
}

__global__ __launch_bounds__(TPB) void sage_gemm_k(const _Float16* __restrict__ X,
                                                   const float* __restrict__ Agg,
                                                   const _Float16* __restrict__ Wr,
                                                   const _Float16* __restrict__ Wl,
                                                   const float* __restrict__ bias,
                                                   float* __restrict__ Out,
                                                   int M, int N, int K) {
  const int lane = threadIdx.x & 31;
  const int wave = threadIdx.x >> 5;
  const int half = lane >> 4;
  const int l16  = lane & 15;
  const int tileM = (blockIdx.x * 8 + wave) * 16;
  const int tileN = blockIdx.y * 64;

  int mA = tileM + l16;
  if (mA >= M) mA = M - 1;   // clamp loads; stores are guarded

  v8f acc[4];
#pragma unroll
  for (int t = 0; t < 4; ++t)
#pragma unroll
    for (int r = 0; r < 8; ++r) acc[t][r] = 0.f;

  for (int k0 = 0; k0 < K; k0 += 32) {
    v16h ax = load_a_f16(X, K, mA, k0, half);
    v16h ag = load_a_f32(Agg, K, mA, k0, half);
#pragma unroll
    for (int t = 0; t < 4; ++t) {
      v16h br = load_b_f16(Wr, K, tileN + t * 16 + l16, k0, half);
      acc[t] = __builtin_amdgcn_wmma_f32_16x16x32_f16(false, ax, false, br,
                                                      (short)0, acc[t], false, false);
      v16h bl = load_b_f16(Wl, K, tileN + t * 16 + l16, k0, half);
      acc[t] = __builtin_amdgcn_wmma_f32_16x16x32_f16(false, ag, false, bl,
                                                      (short)0, acc[t], false, false);
    }
  }
  // C/D layout: VGPR r -> row tileM + r + half*8, col tileN + t*16 + l16
#pragma unroll
  for (int t = 0; t < 4; ++t) {
    int col = tileN + t * 16 + l16;
    float bv = bias[col];
#pragma unroll
    for (int r = 0; r < 8; ++r) {
      int row = tileM + r + half * 8;
      if (row < M) Out[(long long)row * N + col] = acc[t][r] + bv;
    }
  }
}

// ---------------------------------------------------------------- LayerNorm(graph) + ReLU
__global__ __launch_bounds__(TPB) void reduce_stats_k(const float* __restrict__ X,
                                                      long long n, float* stats) {
  __shared__ float ss[TPB];
  __shared__ float sq[TPB];
  float s = 0.f, q = 0.f;
  long long stride = (long long)gridDim.x * blockDim.x;
  for (long long i = (long long)blockIdx.x * blockDim.x + threadIdx.x; i < n; i += stride) {
    float v = X[i]; s += v; q += v * v;
  }
  ss[threadIdx.x] = s; sq[threadIdx.x] = q;
  __syncthreads();
  for (int st = TPB / 2; st > 0; st >>= 1) {
    if ((int)threadIdx.x < st) {
      ss[threadIdx.x] += ss[threadIdx.x + st];
      sq[threadIdx.x] += sq[threadIdx.x + st];
    }
    __syncthreads();
  }
  if (threadIdx.x == 0) { atomicAdd(&stats[0], ss[0]); atomicAdd(&stats[1], sq[0]); }
}

__global__ __launch_bounds__(TPB) void ln_relu_k(const float* __restrict__ Pre,
                                                 const float* __restrict__ stats,
                                                 const float* __restrict__ lnw,
                                                 const float* __restrict__ lnb,
                                                 _Float16* __restrict__ Out,
                                                 long long n, int C) {
  float invn = 1.f / (float)n;
  float mean = stats[0] * invn;
  float var  = stats[1] * invn - mean * mean;
  float isd  = 1.f / (sqrtf(fmaxf(var, 0.f)) + 1e-5f);
  long long stride = (long long)gridDim.x * blockDim.x;
  for (long long i = (long long)blockIdx.x * blockDim.x + threadIdx.x; i < n; i += stride) {
    int c = (int)(i % C);
    float v = (Pre[i] - mean) * isd * lnw[c] + lnb[c];
    Out[i] = (_Float16)fmaxf(v, 0.f);
  }
}

// ---------------------------------------------------------------- pooling + graph FC
__global__ __launch_bounds__(TPB) void pool_add_k(const _Float16* __restrict__ H,
                                                  const int* __restrict__ batch,
                                                  float* __restrict__ pooled,
                                                  float* __restrict__ gcnt,
                                                  long long M, int C) {
  long long n = M * (long long)C;
  long long stride = (long long)gridDim.x * blockDim.x;
  for (long long i = (long long)blockIdx.x * blockDim.x + threadIdx.x; i < n; i += stride) {
    long long m = i / C; int c = (int)(i - m * C);
    int g = batch ? batch[m] : 0;
    atomicAdd(&pooled[(long long)g * C + c], (float)H[i]);
    if (c == 0) atomicAdd(&gcnt[g], 1.f);
  }
}

__global__ __launch_bounds__(128) void graph_fc_k(const float* __restrict__ pooled,
                                                  const float* __restrict__ gcnt,
                                                  const float* __restrict__ mw,
                                                  const float* __restrict__ fcW,
                                                  const float* __restrict__ fcb,
                                                  float* __restrict__ emb,
                                                  int C, int EMBd, int hasMw) {
  int g = blockIdx.x, j = threadIdx.x;
  if (j >= EMBd) return;
  int ind = C + hasMw;
  float cnt = fmaxf(gcnt[g], 1.f);
  const float* wrow = fcW + (long long)j * ind;
  float acc = fcb[j];
  for (int c = 0; c < C; ++c) acc += (pooled[(long long)g * C + c] / cnt) * wrow[c];
  if (hasMw) acc += log10f(mw[g]) * wrow[C];
  emb[(long long)g * EMBd + j] = tanhf(acc);
}

// ---------------------------------------------------------------- attention + MLP tail
__global__ __launch_bounds__(TPB) void finalize_k(const float* __restrict__ emb,      // [16,128]
                                                  const float* __restrict__ salt_emb, // [128]
                                                  const float* __restrict__ frac,     // [16]
                                                  const float* __restrict__ salt_mol, // [1]
                                                  const float* qW, const float* qb,
                                                  const float* kW, const float* kb,
                                                  const float* vW, const float* vb,
                                                  const float* W1, const float* b1,
                                                  const float* W2, const float* b2,
                                                  const float* W3, const float* b3,
                                                  float* out) {
  __shared__ float q[16 * 64], kk[16 * 64], v[16 * 128];
  __shared__ float sm[16 * 16], wj[16], xx[257], h1[256], h2[256];
  int tid = threadIdx.x;
  for (int i = tid; i < 16 * 64; i += TPB) {
    int g = i >> 6, j = i & 63;
    float a = qb[j], b = kb[j];
    for (int c = 0; c < 128; ++c) {
      float e = emb[g * 128 + c];
      a += e * qW[j * 128 + c];
      b += e * kW[j * 128 + c];
    }
    q[i] = a; kk[i] = b;
  }
  for (int i = tid; i < 16 * 128; i += TPB) {
    int g = i >> 7, j = i & 127;
    float a = vb[j];
    for (int c = 0; c < 128; ++c) a += emb[g * 128 + c] * vW[j * 128 + c];
    v[i] = a;
  }
  __syncthreads();
  for (int i = tid; i < 256; i += TPB) {
    int r = i >> 4, c = i & 15;
    float s = 0.f;
    for (int d = 0; d < 64; ++d) s += q[r * 64 + d] * kk[c * 64 + d];
    sm[i] = s * 0.125f;                       // / sqrt(64)
  }
  __syncthreads();
  if (tid < 16) {                             // softmax over axis 0 (per column), fold frac
    int c = tid;
    float mx = -1e30f;
    for (int r = 0; r < 16; ++r) mx = fmaxf(mx, sm[r * 16 + c]);
    float s = 0.f;
    for (int r = 0; r < 16; ++r) { float e = expf(sm[r * 16 + c] - mx); sm[r * 16 + c] = e; s += e; }
    float w = 0.f;
    for (int r = 0; r < 16; ++r) w += frac[r] * (sm[r * 16 + c] / s);
    wj[c] = w;
  }
  __syncthreads();
  for (int d = tid; d < 128; d += TPB) {
    float s = 0.f;
    for (int j = 0; j < 16; ++j) s += wj[j] * v[j * 128 + d];
    xx[d] = s;
    xx[128 + d] = salt_emb[d];
  }
  if (tid == 0) xx[256] = salt_mol[0];
  __syncthreads();
  for (int o = tid; o < 256; o += TPB) {
    float s = b1[o];
    for (int c = 0; c < 257; ++c) s += xx[c] * W1[o * 257 + c];
    h1[o] = fmaxf(s, 0.f);
  }
  __syncthreads();
  for (int o = tid; o < 256; o += TPB) {
    float s = b2[o];
    for (int c = 0; c < 256; ++c) s += h1[c] * W2[o * 256 + c];
    h2[o] = fmaxf(s, 0.f);
  }
  __syncthreads();
  if (tid == 0) {
    float s = b3[0];
    for (int c = 0; c < 256; ++c) s += h2[c] * W3[c];
    out[0] = s;
  }
}

// ---------------------------------------------------------------- launcher
static inline dim3 gsz(long long n) {
  long long b = (n + TPB - 1) / TPB;
  if (b > 16384) b = 16384;
  if (b < 1) b = 1;
  return dim3((unsigned)b);
}

extern "C" void kernel_launch(void* const* d_in, const int* in_sizes, int n_in,
                              void* d_out, int out_size, void* d_ws, size_t ws_size,
                              hipStream_t stream) {
  (void)n_in; (void)out_size; (void)ws_size;
  const long long M  = 200000, Ms = 5000;
  const long long E  = 800000, Es = 20000;
  const int FIN = 32, HID = 256, EMB = 128, NG = 16;

  const float* x        = (const float*)d_in[0];
  const float* mw       = (const float*)d_in[2];
  const float* frac     = (const float*)d_in[3];
  const float* salt_mol = (const float*)d_in[4];
  const float* salt_x   = (const float*)d_in[5];
  const float* P        = (const float*)d_in[7];
  const int*   ei       = (const int*)d_in[8];
  const int*   batch    = (const int*)d_in[9];
  const int*   sei      = (const int*)d_in[10];
  (void)in_sizes;
  const int* srcM = ei;        const int* dstM = ei + E;
  const int* srcS = sei;       const int* dstS = sei + Es;

  // ---- params (jax tree-flatten order: dict keys sorted, tuples positional)
  size_t po = 0;
  auto take = [&](size_t n) { const float* p = P + po; po += n; return p; };
  const float* attk_W = take(64 * 128);  const float* attk_b = take(64);
  const float* attq_W = take(64 * 128);  const float* attq_b = take(64);
  const float* attv_W = take(128 * 128); const float* attv_b = take(128);
  const float* phWl[3]; const float* phWr[3]; const float* phbl[3];
  phWl[0] = take(256 * 32);  phWr[0] = take(256 * 32);  phbl[0] = take(256);
  phWl[1] = take(256 * 256); phWr[1] = take(256 * 256); phbl[1] = take(256);
  phWl[2] = take(256 * 256); phWr[2] = take(256 * 256); phbl[2] = take(256);
  const float* phfcW = take(128 * 257); const float* phfcb = take(128);
  const float* phlnb = take(256);       const float* phlnw = take(256);
  const float* psWl[3]; const float* psWr[3]; const float* psbl[3];
  psWl[0] = take(256 * 32);  psWr[0] = take(256 * 32);  psbl[0] = take(256);
  psWl[1] = take(256 * 256); psWr[1] = take(256 * 256); psbl[1] = take(256);
  psWl[2] = take(256 * 256); psWr[2] = take(256 * 256); psbl[2] = take(256);
  const float* psfcW = take(128 * 256); const float* psfcb = take(128);
  const float* pslnb = take(256);       const float* pslnw = take(256);
  const float* r1W = take(256 * 257); const float* r1b = take(256);
  const float* r2W = take(256 * 256); const float* r2b = take(256);
  const float* r3W = take(256);       const float* r3b = take(1);

  // ---- workspace carve-out
  char* ws = (char*)d_ws;
  size_t off = 0;
  auto alloc = [&](size_t bytes) { void* p = ws + off; off = (off + bytes + 255) & ~(size_t)255; return p; };
  float*     cnt    = (float*)alloc(M * 4);
  float*     cntS   = (float*)alloc(Ms * 4);
  _Float16*  xh     = (_Float16*)alloc(M * FIN * 2);
  _Float16*  xhS    = (_Float16*)alloc(Ms * FIN * 2);
  _Float16*  h16    = (_Float16*)alloc(M * HID * 2);
  _Float16*  h16S   = (_Float16*)alloc(Ms * HID * 2);
  float*     agg    = (float*)alloc(M * HID * 4);
  float*     aggS   = (float*)alloc(Ms * HID * 4);
  float*     pre    = (float*)alloc(M * HID * 4);
  float*     preS   = (float*)alloc(Ms * HID * 4);
  float*     stats  = (float*)alloc(8 * 4);
  float*     pooled = (float*)alloc(NG * HID * 4);
  float*     gcnt   = (float*)alloc(NG * 4);
  float*     pooledS = (float*)alloc(HID * 4);
  float*     gcntS  = (float*)alloc(4);
  float*     emb    = (float*)alloc(NG * EMB * 4);
  float*     embS   = (float*)alloc(EMB * 4);
  _Float16* fWl[3], *fWr[3], *sWl[3], *sWr[3];
  const size_t wsz[3] = {256 * 32, 256 * 256, 256 * 256};
  for (int l = 0; l < 3; ++l) {
    fWl[l] = (_Float16*)alloc(wsz[l] * 2); fWr[l] = (_Float16*)alloc(wsz[l] * 2);
    sWl[l] = (_Float16*)alloc(wsz[l] * 2); sWr[l] = (_Float16*)alloc(wsz[l] * 2);
  }

  // ---- degrees + conversions
  zero_f32_k<<<gsz(M), TPB, 0, stream>>>(cnt, M);
  zero_f32_k<<<gsz(Ms), TPB, 0, stream>>>(cntS, Ms);
  degree_k<<<gsz(E), TPB, 0, stream>>>(dstM, cnt, E);
  degree_k<<<gsz(Es), TPB, 0, stream>>>(dstS, cntS, Es);
  cvt_f32_f16_k<<<gsz(M * FIN), TPB, 0, stream>>>(x, xh, M * FIN);
  cvt_f32_f16_k<<<gsz(Ms * FIN), TPB, 0, stream>>>(salt_x, xhS, Ms * FIN);
  for (int l = 0; l < 3; ++l) {
    cvt_f32_f16_k<<<gsz(wsz[l]), TPB, 0, stream>>>(phWl[l], fWl[l], (long long)wsz[l]);
    cvt_f32_f16_k<<<gsz(wsz[l]), TPB, 0, stream>>>(phWr[l], fWr[l], (long long)wsz[l]);
    cvt_f32_f16_k<<<gsz(wsz[l]), TPB, 0, stream>>>(psWl[l], sWl[l], (long long)wsz[l]);
    cvt_f32_f16_k<<<gsz(wsz[l]), TPB, 0, stream>>>(psWr[l], sWr[l], (long long)wsz[l]);
  }

  // ---- main graph: 3 SAGE layers
  for (int l = 0; l < 3; ++l) {
    int K = l ? HID : FIN;
    const _Float16* Xin = l ? h16 : xh;
    zero_f32_k<<<gsz(M * K), TPB, 0, stream>>>(agg, M * K);
    if (l == 0) scatter_f32_k<<<gsz(E * K), TPB, 0, stream>>>(x, srcM, dstM, agg, E, K);
    else        scatter_f16_k<<<gsz(E * K), TPB, 0, stream>>>(h16, srcM, dstM, agg, E, K);
    agg_div_k<<<gsz(M * K), TPB, 0, stream>>>(agg, cnt, M, K);
    dim3 gg((unsigned)((M + 127) / 128), HID / 64);
    sage_gemm_k<<<gg, TPB, 0, stream>>>(Xin, agg, fWr[l], fWl[l], phbl[l], pre, (int)M, HID, K);
    zero_f32_k<<<1, TPB, 0, stream>>>(stats, 2);
    reduce_stats_k<<<1024, TPB, 0, stream>>>(pre, M * HID, stats);
    ln_relu_k<<<gsz(M * HID), TPB, 0, stream>>>(pre, stats, phlnw, phlnb, h16, M * HID, HID);
  }
  zero_f32_k<<<gsz(NG * HID), TPB, 0, stream>>>(pooled, NG * HID);
  zero_f32_k<<<1, TPB, 0, stream>>>(gcnt, NG);
  pool_add_k<<<gsz(M * HID), TPB, 0, stream>>>(h16, batch, pooled, gcnt, M, HID);
  graph_fc_k<<<NG, 128, 0, stream>>>(pooled, gcnt, mw, phfcW, phfcb, emb, HID, EMB, 1);

  // ---- salt graph: 3 SAGE layers (batch == all zeros)
  for (int l = 0; l < 3; ++l) {
    int K = l ? HID : FIN;
    const _Float16* Xin = l ? h16S : xhS;
    zero_f32_k<<<gsz(Ms * K), TPB, 0, stream>>>(aggS, Ms * K);
    if (l == 0) scatter_f32_k<<<gsz(Es * K), TPB, 0, stream>>>(salt_x, srcS, dstS, aggS, Es, K);
    else        scatter_f16_k<<<gsz(Es * K), TPB, 0, stream>>>(h16S, srcS, dstS, aggS, Es, K);
    agg_div_k<<<gsz(Ms * K), TPB, 0, stream>>>(aggS, cntS, Ms, K);
    dim3 gg((unsigned)((Ms + 127) / 128), HID / 64);
    sage_gemm_k<<<gg, TPB, 0, stream>>>(Xin, aggS, sWr[l], sWl[l], psbl[l], preS, (int)Ms, HID, K);
    zero_f32_k<<<1, TPB, 0, stream>>>(stats, 2);
    reduce_stats_k<<<256, TPB, 0, stream>>>(preS, Ms * HID, stats);
    ln_relu_k<<<gsz(Ms * HID), TPB, 0, stream>>>(preS, stats, pslnw, pslnb, h16S, Ms * HID, HID);
  }
  zero_f32_k<<<1, TPB, 0, stream>>>(pooledS, HID);
  zero_f32_k<<<1, TPB, 0, stream>>>(gcntS, 1);
  pool_add_k<<<gsz(Ms * HID), TPB, 0, stream>>>(h16S, (const int*)nullptr, pooledS, gcntS, Ms, HID);
  graph_fc_k<<<1, 128, 0, stream>>>(pooledS, gcntS, (const float*)nullptr, psfcW, psfcb, embS, HID, EMB, 0);

  // ---- attention + MLP tail
  finalize_k<<<1, TPB, 0, stream>>>(emb, embS, frac, salt_mol,
                                    attq_W, attq_b, attk_W, attk_b, attv_W, attv_b,
                                    r1W, r1b, r2W, r2b, r3W, r3b, (float*)d_out);
}
